// SuperGlueAttentionalGNN_19215683682998
// MI455X (gfx1250) — matrix operand order repaired and checked by reading, not compiled, not run on hardware
//
#include <hip/hip_runtime.h>
#include <hip/hip_bf16.h>

typedef __attribute__((ext_vector_type(16))) _Float16 v16h;
typedef __attribute__((ext_vector_type(8)))  float    v8f;

union V16 { v16h v; uint4 u[2]; };

#define LL 18
#define DD 256
#define NN 1024
#define BB 4

__device__ __forceinline__ v8f wmma_f16(v16h a, v16h b, v8f c) {
  return __builtin_amdgcn_wmma_f32_16x16x32_f16(false, a, false, b, (short)0, c, false, false);
}

// A fragment (16x32 f16): lane lo = row M; e<8 -> K=hi*8+e ; e>=8 -> K=16+hi*8+(e-8)
__device__ __forceinline__ v16h frag_a(const _Float16* rowptr, int hi) {
  V16 a;
  a.u[0] = *(const uint4*)(rowptr + hi * 8);
  a.u[1] = *(const uint4*)(rowptr + 16 + hi * 8);
  return a.v;
}

// B fragment (32x16 f16): lane lo = column N; e<8 -> K=hi*8+e ; e>=8 -> K=16+hi*8+(e-8)
__device__ __forceinline__ v16h frag_b(const _Float16* rowptr, int hi) {
  V16 b;
  b.u[0] = *(const uint4*)(rowptr + hi * 8);
  b.u[1] = *(const uint4*)(rowptr + 16 + hi * 8);
  return b.v;
}

// CDNA5 async global->LDS DMA (tracked by ASYNCcnt; completes in order)
__device__ __forceinline__ void async_copy_b128(void* lds_dst, const void* gsrc) {
  unsigned lds = (unsigned)(uintptr_t)lds_dst;
  unsigned long long ga = (unsigned long long)(uintptr_t)gsrc;
  asm volatile("global_load_async_to_lds_b128 %0, %1, off" :: "v"(lds), "v"(ga) : "memory");
}
__device__ __forceinline__ void wait_async_le0() {
  asm volatile("s_wait_asynccnt 0x0" ::: "memory");
}
__device__ __forceinline__ void wait_async_le3() {  // keep next tile's 3 DMAs in flight
  asm volatile("s_wait_asynccnt 0x3" ::: "memory");
}

// ---------------------------------------------------------------------------
// Token-major WMMA GEMM with double-buffered async-DMA staging:
//   A = activations X_t (tokens, K) token-major (per-batch strideA, ld = lda)
//   B = weights W (Mout, K) row-major (as stored)
//   D fragment: tokens in VGPRs (M), out-channels on lanes (N)
// Block: 64 tokens x 128 out-channels, K step 32, 256 thr = 8 waves (2x4).
// MODE 0: +bias             -> f16 token-major out
// MODE 1: *scale+shift,relu -> f16 token-major out      (folded BatchNorm)
// MODE 2: +bias +residual   -> f32 channel-major (vectorized) + f16 token-major
// MODE 3: +bias             -> f16 CHANNEL-major out (contiguous b128 stores)
// ---------------------------------------------------------------------------
template <int MODE>
__global__ __launch_bounds__(256) void gemm_tn(
    const _Float16* __restrict__ gX, long long strideA, int lda, int K,
    const _Float16* __restrict__ gW,
    const float* __restrict__ vec0, const float* __restrict__ vec1,
    _Float16* __restrict__ gOh, long long strideOh, int ldoh,
    float* __restrict__ gOf, long long strideOf,
    int cross, int swapFrom) {
  __shared__ __align__(16) _Float16 ldsX[2][64 * 32];
  __shared__ __align__(16) _Float16 ldsW[2][128 * 32];

  const int t = threadIdx.x;
  const int lane = t & 31, w = t >> 5;
  const int lo = lane & 15, hi = lane >> 4;
  const int wt = w >> 2, wo = w & 3;      // wave tile: token half x outch quarter
  const int o0 = blockIdx.x * 128;        // out-channel tile
  const int t0 = blockIdx.y * 64;         // token tile
  const int bz = blockIdx.z;
  const int bsrc = (cross && o0 >= swapFrom) ? (bz ^ 2) : bz;
  const _Float16* A = gX + (size_t)bsrc * (size_t)strideA;

  // per-thread staging coordinates (3 async b128 per thread per K-tile)
  const int sxN = t >> 2, sxK = (t & 3) * 8;          // X tile [64][32]
  const int swO0 = (t * 2) >> 2, swK0 = ((t * 2) & 3) * 8;
  const int swO1 = (t * 2 + 1) >> 2, swK1 = ((t * 2 + 1) & 3) * 8;

  auto stage = [&](int buf, int k0) {
    async_copy_b128(&ldsX[buf][sxN * 32 + sxK], A + (size_t)(t0 + sxN) * lda + k0 + sxK);
    async_copy_b128(&ldsW[buf][swO0 * 32 + swK0], gW + (size_t)(o0 + swO0) * K + k0 + swK0);
    async_copy_b128(&ldsW[buf][swO1 * 32 + swK1], gW + (size_t)(o0 + swO1) * K + k0 + swK1);
  };

  v8f acc[2][2];
#pragma unroll
  for (int i = 0; i < 2; ++i)
#pragma unroll
    for (int j = 0; j < 2; ++j)
#pragma unroll
      for (int r = 0; r < 8; ++r) acc[i][j][r] = 0.0f;

  const int nk = K >> 5;
  stage(0, 0);
  for (int ki = 0; ki < nk; ++ki) {
    const int cur = ki & 1;
    __syncthreads();  // everyone done computing from buf cur^1 -> safe to refill
    if (ki + 1 < nk) {
      stage(cur ^ 1, (ki + 1) << 5);
      wait_async_le3();  // drain current tile's 3 DMAs; next tile stays in flight
    } else {
      wait_async_le0();
    }
    __syncthreads();  // all waves' DMAs for buf cur have landed

    v16h afr[2], bfr[2];
#pragma unroll
    for (int ft = 0; ft < 2; ++ft)
      afr[ft] = frag_a(&ldsX[cur][(wt * 32 + ft * 16 + lo) * 32], hi);
#pragma unroll
    for (int fo = 0; fo < 2; ++fo)
      bfr[fo] = frag_b(&ldsW[cur][(wo * 32 + fo * 16 + lo) * 32], hi);
#pragma unroll
    for (int ft = 0; ft < 2; ++ft)
#pragma unroll
      for (int fo = 0; fo < 2; ++fo)
        acc[ft][fo] = wmma_f16(afr[ft], bfr[fo], acc[ft][fo]);
  }

  // epilogue: tokens tb..tb+7 in VGPRs, out-channel och on lanes
#pragma unroll
  for (int ft = 0; ft < 2; ++ft)
#pragma unroll
    for (int fo = 0; fo < 2; ++fo) {
      const int och = o0 + wo * 32 + fo * 16 + lo;
      const int tb = t0 + wt * 32 + ft * 16 + hi * 8;
      if (MODE == 0 || MODE == 1) {
        float b0 = vec0[och];
        float b1 = (MODE == 1) ? vec1[och] : 0.0f;
#pragma unroll
        for (int r = 0; r < 8; ++r) {
          float v = acc[ft][fo][r];
          v = (MODE == 1) ? fmaxf(v * b0 + b1, 0.0f) : (v + b0);
          gOh[(size_t)bz * strideOh + (size_t)(tb + r) * ldoh + och] = (_Float16)v;
        }
      } else if (MODE == 3) {  // channel-major: one b128 store
        float b0 = vec0[och];
        union { uint4 u; _Float16 h[8]; } pk;
#pragma unroll
        for (int r = 0; r < 8; ++r) pk.h[r] = (_Float16)(acc[ft][fo][r] + b0);
        *(uint4*)(gOh + (size_t)bz * strideOh + (size_t)och * ldoh + tb) = pk.u;
      } else {  // MODE 2: residual, vectorized f32 channel-major + f16 token-major
        float b0 = vec0[och];
        float* xp = gOf + (size_t)bz * strideOf + (size_t)och * NN + tb;
        float4 x0 = *(float4*)xp, x1 = *(float4*)(xp + 4);
        float v[8];
        v[0] = acc[ft][fo][0] + b0 + x0.x; v[1] = acc[ft][fo][1] + b0 + x0.y;
        v[2] = acc[ft][fo][2] + b0 + x0.z; v[3] = acc[ft][fo][3] + b0 + x0.w;
        v[4] = acc[ft][fo][4] + b0 + x1.x; v[5] = acc[ft][fo][5] + b0 + x1.y;
        v[6] = acc[ft][fo][6] + b0 + x1.z; v[7] = acc[ft][fo][7] + b0 + x1.w;
        *(float4*)xp = make_float4(v[0], v[1], v[2], v[3]);
        *(float4*)(xp + 4) = make_float4(v[4], v[5], v[6], v[7]);
#pragma unroll
        for (int r = 0; r < 8; ++r)
          gOh[(size_t)bz * strideOh + (size_t)(tb + r) * ldoh + och] = (_Float16)v[r];
      }
    }
}

// ---------------------------------------------------------------------------
// Flash attention, per head. No LDS, no barriers. 128 thr = 4 waves x 16 queries.
//   QKt: (b, 1024, 512) token-major, cols [0,256)=Q, [256,512)=K (head-contig)
//   Vc : (b, 256, 1024) channel-major (head-contig rows)
//   Ot : (b, 1024, 256) token-major
// Scores S^T = K^T Q: lane = query, keys in VGPRs (in-lane softmax + xor16).
// The S^T fragment IS the A-operand layout for P in O = P x V: no shuffles.
// ---------------------------------------------------------------------------
__global__ __launch_bounds__(128) void attn_kernel(const _Float16* __restrict__ QKt,
                                                   const _Float16* __restrict__ Vc,
                                                   _Float16* __restrict__ Ot) {
  const int t = threadIdx.x;
  const int lane = t & 31, w = t >> 5;
  const int lo = lane & 15, hi = lane >> 4;
  const int b = blockIdx.y >> 2, h = blockIdx.y & 3;
  const int q0 = blockIdx.x * 64 + w * 16;  // this wave's 16 queries

  const _Float16* Qb = QKt + (size_t)b * NN * 512 + h * 64;
  const _Float16* Kb = Qb + 256;
  const _Float16* Vb = Vc + (size_t)b * 256 * NN + (size_t)h * 64 * NN;

  // Q B-operand fragments: loop-invariant, preload (lane = query column)
  v16h bq[2];
#pragma unroll
  for (int j = 0; j < 2; ++j)
    bq[j] = frag_b(Qb + (size_t)(q0 + lo) * 512 + j * 32, hi);

  float Mrow = -1e30f, Lrow = 0.0f;
  v8f Of[4];
#pragma unroll
  for (int cf = 0; cf < 4; ++cf)
#pragma unroll
    for (int r = 0; r < 8; ++r) Of[cf][r] = 0.0f;

  const float cs = 0.125f * 1.44269504088896f;  // 1/sqrt(64) * log2(e)

  for (int m0 = 0; m0 < NN; m0 += 32) {
    v8f s0, s1;
#pragma unroll
    for (int r = 0; r < 8; ++r) { s0[r] = 0.0f; s1[r] = 0.0f; }
#pragma unroll
    for (int j = 0; j < 2; ++j) {  // channel chunks 0..31, 32..63
      v16h a0 = frag_a(Kb + (size_t)(m0 + lo) * 512 + j * 32, hi);       // keys m0+0..15
      v16h a1 = frag_a(Kb + (size_t)(m0 + 16 + lo) * 512 + j * 32, hi);  // keys m0+16..31
      s0 = wmma_f16(a0, bq[j], s0);
      s1 = wmma_f16(a1, bq[j], s1);
    }

    // online softmax over keys: 16 in-lane + xor16 across lane halves
    float ml = -1e30f;
#pragma unroll
    for (int r = 0; r < 8; ++r) {
      s0[r] *= cs; s1[r] *= cs;
      ml = fmaxf(ml, fmaxf(s0[r], s1[r]));
    }
    ml = fmaxf(ml, __shfl_xor(ml, 16));
    float Mn = fmaxf(Mrow, ml);
    float alpha = exp2f(Mrow - Mn);
    float p0[8], p1[8], lsum = 0.0f;
#pragma unroll
    for (int r = 0; r < 8; ++r) {
      p0[r] = exp2f(s0[r] - Mn);
      p1[r] = exp2f(s1[r] - Mn);
      lsum += p0[r] + p1[r];
    }
    lsum += __shfl_xor(lsum, 16);
    Lrow = Lrow * alpha + lsum;
    Mrow = Mn;
#pragma unroll
    for (int cf = 0; cf < 4; ++cf)
#pragma unroll
      for (int r = 0; r < 8; ++r) Of[cf][r] *= alpha;

    // P is already in A-operand layout: ap[e] = p0[e], ap[8+e] = p1[e]
    v16h ap;
#pragma unroll
    for (int r = 0; r < 8; ++r) {
      ap[r] = (_Float16)p0[r];
      ap[8 + r] = (_Float16)p1[r];
    }
    // O += P x V ; V B-operand direct from channel-major global (key-contig)
#pragma unroll
    for (int cf = 0; cf < 4; ++cf) {
      v16h bv = frag_b(Vb + (size_t)(cf * 16 + lo) * NN + m0, hi);
      Of[cf] = wmma_f16(ap, bv, Of[cf]);
    }
  }

  const float inv = 1.0f / Lrow;
  _Float16* Ob = Ot + (size_t)b * NN * 256 + h * 64;
#pragma unroll
  for (int cf = 0; cf < 4; ++cf)
#pragma unroll
    for (int r = 0; r < 8; ++r)
      Ob[(size_t)(q0 + r + hi * 8) * 256 + cf * 16 + lo] = (_Float16)(Of[cf][r] * inv);
}

// ---------------------------------------------------------------------------
// Conversion / prep kernels
// ---------------------------------------------------------------------------
__global__ void k_conv_qkv(const float* __restrict__ pw, _Float16* __restrict__ wq) {
  long long idx = (long long)blockIdx.x * blockDim.x + threadIdx.x;
  if (idx >= (long long)LL * 3 * 256 * 256) return;
  int i = (int)(idx & 255);
  long long row = idx >> 8;           // l*768 + sec*256 + op
  int op = (int)(row & 255);
  int o = (op & 63) * 4 + (op >> 6);  // (h,hd) -> original hd*H+h
  wq[idx] = (_Float16)pw[(row - op + o) * 256 + i];
}

__global__ void k_conv_merge(const float* __restrict__ mw, _Float16* __restrict__ wm) {
  long long idx = (long long)blockIdx.x * blockDim.x + threadIdx.x;
  if (idx >= (long long)LL * 256 * 256) return;
  int cp = (int)(idx & 255);
  long long ro = idx >> 8;
  int c = (cp & 63) * 4 + (cp >> 6);  // column permutation
  wm[idx] = (_Float16)mw[ro * 256 + c];
}

__global__ void k_conv_plain(const float* __restrict__ src, _Float16* __restrict__ dst,
                             long long count) {
  long long idx = (long long)blockIdx.x * blockDim.x + threadIdx.x;
  if (idx < count) dst[idx] = (_Float16)src[idx];
}

__global__ void k_prep_vec(const float* __restrict__ pb, const float* __restrict__ mb,
                           const float* __restrict__ b1, const float* __restrict__ g,
                           const float* __restrict__ be, const float* __restrict__ mn,
                           const float* __restrict__ vr, const float* __restrict__ b2,
                           float* __restrict__ BQKV, float* __restrict__ BM,
                           float* __restrict__ S1, float* __restrict__ T1,
                           float* __restrict__ B2) {
  int idx = blockIdx.x * blockDim.x + threadIdx.x;
  if (idx < LL * 768) {
    int op = idx & 255;
    int o = (op & 63) * 4 + (op >> 6);
    BQKV[idx] = pb[idx - op + o];
  }
  if (idx < LL * 512) {
    float s = g[idx] * rsqrtf(vr[idx] + 1e-5f);
    S1[idx] = s;
    T1[idx] = (b1[idx] - mn[idx]) * s + be[idx];
  }
  if (idx < LL * 256) {
    BM[idx] = mb[idx];
    B2[idx] = b2[idx];
  }
}

// token-major f16 copy of x (transpose of channel-major inputs)
__global__ void k_init_xh(const float* __restrict__ d0, const float* __restrict__ d1,
                          _Float16* __restrict__ XH) {
  long long idx = (long long)blockIdx.x * blockDim.x + threadIdx.x;
  if (idx >= (long long)BB * DD * NN) return;
  int b = (int)(idx >> 18);
  int m = (int)((idx >> 10) & 255);
  int n = (int)(idx & 1023);
  float v = (b < 2) ? d0[(size_t)b * DD * NN + m * NN + n]
                    : d1[(size_t)(b - 2) * DD * NN + m * NN + n];
  XH[(size_t)b * NN * 512 + (size_t)n * 512 + m] = (_Float16)v;
}

// ---------------------------------------------------------------------------
extern "C" void kernel_launch(void* const* d_in, const int* in_sizes, int n_in,
                              void* d_out, int out_size, void* d_ws, size_t ws_size,
                              hipStream_t stream) {
  (void)in_sizes; (void)n_in; (void)out_size; (void)ws_size;
  const float* d0v     = (const float*)d_in[0];
  const float* d1v     = (const float*)d_in[1];
  const float* proj_w  = (const float*)d_in[2];
  const float* proj_b  = (const float*)d_in[3];
  const float* merge_w = (const float*)d_in[4];
  const float* merge_b = (const float*)d_in[5];
  const float* mlp_w1  = (const float*)d_in[6];
  const float* mlp_b1  = (const float*)d_in[7];
  const float* bn_g    = (const float*)d_in[8];
  const float* bn_b    = (const float*)d_in[9];
  const float* bn_m    = (const float*)d_in[10];
  const float* bn_v    = (const float*)d_in[11];
  const float* mlp_w2  = (const float*)d_in[12];
  const float* mlp_b2  = (const float*)d_in[13];

  char* ws = (char*)d_ws;
  size_t off = 0;
  auto carve = [&](size_t bytes) -> void* {
    void* p = ws + off;
    off += (bytes + 255) & ~(size_t)255;
    return p;
  };
  _Float16* WQKV = (_Float16*)carve((size_t)LL * 768 * 256 * 2);
  _Float16* WM   = (_Float16*)carve((size_t)LL * 256 * 256 * 2);
  _Float16* W1   = (_Float16*)carve((size_t)LL * 512 * 512 * 2);
  _Float16* W2   = (_Float16*)carve((size_t)LL * 256 * 512 * 2);
  float* BQKV    = (float*)carve((size_t)LL * 768 * 4);
  float* BM      = (float*)carve((size_t)LL * 256 * 4);
  float* S1      = (float*)carve((size_t)LL * 512 * 4);
  float* T1      = (float*)carve((size_t)LL * 512 * 4);
  float* B2      = (float*)carve((size_t)LL * 256 * 4);
  float* Xf32    = (float*)carve((size_t)BB * DD * NN * 4);      // channel-major (== d_out layout)
  _Float16* XH   = (_Float16*)carve((size_t)BB * NN * 512 * 2);  // token-major; cols[0,256)=x cols[256,512)=att
  _Float16* QKt  = (_Float16*)carve((size_t)BB * NN * 512 * 2);  // token-major; cols[0,256)=Q cols[256,512)=K
  _Float16* Vc   = (_Float16*)carve((size_t)BB * 256 * NN * 2);  // channel-major
  _Float16* Ot   = (_Float16*)carve((size_t)BB * NN * 256 * 2);  // token-major
  _Float16* H1   = (_Float16*)carve((size_t)BB * NN * 512 * 2);  // token-major

  // prep (every call; deterministic)
  k_conv_qkv<<<(LL * 3 * 256 * 256 + 255) / 256, 256, 0, stream>>>(proj_w, WQKV);
  k_conv_merge<<<(LL * 256 * 256 + 255) / 256, 256, 0, stream>>>(merge_w, WM);
  k_conv_plain<<<((long long)LL * 512 * 512 + 255) / 256, 256, 0, stream>>>(
      mlp_w1, W1, (long long)LL * 512 * 512);
  k_conv_plain<<<((long long)LL * 256 * 512 + 255) / 256, 256, 0, stream>>>(
      mlp_w2, W2, (long long)LL * 256 * 512);
  k_prep_vec<<<(LL * 768 + 255) / 256, 256, 0, stream>>>(
      proj_b, merge_b, mlp_b1, bn_g, bn_b, bn_m, bn_v, mlp_b2, BQKV, BM, S1, T1, B2);
  hipMemcpyAsync(Xf32, d0v, (size_t)2 * DD * NN * 4, hipMemcpyDeviceToDevice, stream);
  hipMemcpyAsync(Xf32 + 2 * DD * NN, d1v, (size_t)2 * DD * NN * 4,
                 hipMemcpyDeviceToDevice, stream);
  k_init_xh<<<(BB * DD * NN + 255) / 256, 256, 0, stream>>>(d0v, d1v, XH);

  const long long sXH = (long long)NN * 512;
  const long long sQK = (long long)NN * 512;
  const long long sVc = 256LL * NN;
  const long long sOt = (long long)NN * 256;
  const long long sX  = 256LL * NN;

  for (int i = 0; i < LL; ++i) {
    int cross = i & 1;  // LAYER_TYPES = ('self','cross')*9
    // Q|K projection (token-major out); K half reads swapped batches when cross
    gemm_tn<0><<<dim3(4, 16, 4), 256, 0, stream>>>(
        XH, sXH, 512, 256, WQKV + (size_t)i * 768 * 256,
        BQKV + i * 768, nullptr, QKt, sQK, 512, nullptr, 0, cross, 256);
    // V projection (channel-major out, vectorized stores); always swapped when cross
    gemm_tn<3><<<dim3(2, 16, 4), 256, 0, stream>>>(
        XH, sXH, 512, 256, WQKV + (size_t)i * 768 * 256 + 512 * 256,
        BQKV + i * 768 + 512, nullptr, Vc, sVc, NN, nullptr, 0, cross, 0);
    // attention (flash, LDS-free)
    attn_kernel<<<dim3(16, 16), 128, 0, stream>>>(QKt, Vc, Ot);
    // merge -> att columns of XH
    gemm_tn<0><<<dim3(2, 16, 4), 256, 0, stream>>>(
        Ot, sOt, 256, 256, WM + (size_t)i * 256 * 256,
        BM + i * 256, nullptr, XH + 256, sXH, 512, nullptr, 0, 0, 0);
    // MLP layer 1 (+folded BN + ReLU)
    gemm_tn<1><<<dim3(4, 16, 4), 256, 0, stream>>>(
        XH, sXH, 512, 512, W1 + (size_t)i * 512 * 512,
        S1 + i * 512, T1 + i * 512, H1, sXH, 512, nullptr, 0, 0, 0);
    // MLP layer 2 (+bias +residual) -> Xf32 (channel-major) and XH x-columns
    gemm_tn<2><<<dim3(2, 16, 4), 256, 0, stream>>>(
        H1, sXH, 512, 512, W2 + (size_t)i * 256 * 512,
        B2 + i * 256, nullptr, XH, sXH, 512, Xf32, sX, 0, 0);
  }

  hipMemcpyAsync(d_out, Xf32, (size_t)BB * DD * NN * sizeof(float),
                 hipMemcpyDeviceToDevice, stream);
}
